// FlaxT5Attention_5566277616070
// MI455X (gfx1250) — matrix operand order repaired and checked by reading, not compiled, not run on hardware
//
#include <hip/hip_runtime.h>
#include <hip/hip_bf16.h>
#include <math.h>

// ---------------------------------------------------------------------------
// T5 encoder self-attention for MI455X (gfx1250), wave32 + WMMA bf16.
// B=4, S=2048, D_MODEL=1024, H=16, D_KV=64.
//
// Pipeline:
//   0a) cvt_x:   hidden_states f32 -> bf16 (one-time)
//   0b) cvt_wt:  Wq/Wk/Wv/Wo f32 [K][N] -> bf16 [N][K] (LDS-tiled transpose)
//   1)  qkv_proj: bf16 WMMA GEMM; Q*0.125 & K -> [B,H,S,D], V -> [B,H,D,S]
//   2)  bias_lut: 4095x16 bucket LUT (replaces ~1G logf calls)
//   3)  bias_out: 1.07 GB position_bias write (pure bandwidth, float4)
//   4)  attn:     flash attention, bf16 WMMA, online softmax
//   5)  out_proj: bf16 WMMA GEMM -> f32 attn_output
// ---------------------------------------------------------------------------

typedef __attribute__((ext_vector_type(16))) __bf16 v16bf;
typedef __attribute__((ext_vector_type(8)))  __bf16 bf16x8;
typedef __attribute__((ext_vector_type(8)))  float  v8f;

#define WMMA_BF16(A, Bm, C) \
    __builtin_amdgcn_wmma_f32_16x16x32_bf16(false, (A), false, (Bm), (short)0, (C), false, false)

constexpr int  B_   = 4;
constexpr int  S_   = 2048;
constexpr int  DM   = 1024;    // d_model
constexpr int  H_   = 16;      // heads
constexpr int  DK   = 64;      // head dim
constexpr int  LUTW = 4096;    // LUT row stride (4095 valid rel positions)

// ---- fragment loaders ------------------------------------------------------
// A-matrix 16x32 bf16: lane holds one row (M = lane&15).
// lanes 0-15: elems 0..15 -> K=0..15
// lanes16-31: elems 0..15 -> K=16..31
__device__ inline v16bf load_a_bf16(const __bf16* __restrict__ row, int kb, int lane) {
    const int klo = (lane & 16) ? 16 : 0;
    const __bf16* p = row + kb + klo;
    bf16x8 lo = *(const bf16x8*)(p);
    bf16x8 hi = *(const bf16x8*)(p + 8);
    v16bf A;
#pragma unroll
    for (int i = 0; i < 8; ++i) { A[i] = lo[i]; A[i + 8] = hi[i]; }
    return A;
}

// B-matrix 32x16 bf16: col N = lane&15; elem i -> K = (lane&16 ? 16 : 0) + i.
// base points at 16 consecutive K elements.
__device__ inline v16bf load_b_bf16c(const __bf16* __restrict__ base) {
    bf16x8 lo = *(const bf16x8*)(base);
    bf16x8 hi = *(const bf16x8*)(base + 8);
    v16bf Bm;
#pragma unroll
    for (int i = 0; i < 8; ++i) { Bm[i] = lo[i]; Bm[i + 8] = hi[i]; }
    return Bm;
}

// ---------------------------------------------------------------------------
// Kernel 0a: f32 -> bf16 straight convert (hidden_states).
// ---------------------------------------------------------------------------
__global__ __launch_bounds__(256)
void cvt_x_kernel(const float* __restrict__ X, __bf16* __restrict__ Xb) {
    const int i = (blockIdx.x * 256 + threadIdx.x) * 8;   // 8 elems/thread
    float4 a = *(const float4*)(X + i);
    float4 b = *(const float4*)(X + i + 4);
    bf16x8 o;
    o[0] = (__bf16)a.x; o[1] = (__bf16)a.y; o[2] = (__bf16)a.z; o[3] = (__bf16)a.w;
    o[4] = (__bf16)b.x; o[5] = (__bf16)b.y; o[6] = (__bf16)b.z; o[7] = (__bf16)b.w;
    *(bf16x8*)(Xb + i) = o;
}

// ---------------------------------------------------------------------------
// Kernel 0b: weight convert + transpose.  Wt[n][k] = (bf16) W[k][n].
// 32x32 LDS tile; blockIdx.z selects which of the 4 weight matrices.
// ---------------------------------------------------------------------------
__global__ __launch_bounds__(256)
void cvt_wt_kernel(const float* __restrict__ Wq, const float* __restrict__ Wk,
                   const float* __restrict__ Wv, const float* __restrict__ Wo,
                   __bf16* __restrict__ Wqt, __bf16* __restrict__ Wkt,
                   __bf16* __restrict__ Wvt, __bf16* __restrict__ Wot) {
    __shared__ float tile[32][33];
    const int z = blockIdx.z;
    const float*  W  = (z == 0) ? Wq  : (z == 1) ? Wk  : (z == 2) ? Wv  : Wo;
    __bf16*       Wt = (z == 0) ? Wqt : (z == 1) ? Wkt : (z == 2) ? Wvt : Wot;
    const int k0 = blockIdx.x * 32, n0 = blockIdx.y * 32;
    const int tx = threadIdx.x & 31, ty = threadIdx.x >> 5;   // 32 x 8
#pragma unroll
    for (int r = ty; r < 32; r += 8)
        tile[r][tx] = W[(size_t)(k0 + r) * DM + n0 + tx];
    __syncthreads();
#pragma unroll
    for (int r = ty; r < 32; r += 8)
        Wt[(size_t)(n0 + r) * DM + k0 + tx] = (__bf16)tile[tx][r];
}

// ---------------------------------------------------------------------------
// Kernel 1: fused QKV projection.  Y = Xb @ W{q,k,v}  (blockIdx.z selects W).
// Xb: [B*S, DM] bf16.  Wt: [N][K] bf16.  Each wave computes 32x64 (2 A-frags
// share each B-frag -> 8 WMMA per K-step).  WG = 4 waves = 128x64 tile.
// ---------------------------------------------------------------------------
__global__ __launch_bounds__(128)
void qkv_proj_kernel(const __bf16* __restrict__ Xb,
                     const __bf16* __restrict__ Wqt,
                     const __bf16* __restrict__ Wkt,
                     const __bf16* __restrict__ Wvt,
                     __bf16* __restrict__ qs,
                     __bf16* __restrict__ kk,
                     __bf16* __restrict__ vt) {
    const int lane  = threadIdx.x & 31;
    const int wave  = threadIdx.x >> 5;
    const int which = blockIdx.z;
    const __bf16* Wt = (which == 0) ? Wqt : (which == 1) ? Wkt : Wvt;

    const int rowBase = blockIdx.x * 128 + wave * 32;
    const int nBase   = blockIdx.y * 64;
    const int khSel   = (lane & 16) ? 16 : 0;

    v8f acc[2][4] = {};
    const __bf16* arow0 = Xb + (size_t)(rowBase + (lane & 15)) * DM;
    const __bf16* arow1 = arow0 + (size_t)16 * DM;

    for (int kb = 0; kb < DM; kb += 32) {
        if (kb + 256 < DM) {
            __builtin_prefetch(arow0 + kb + 256, 0, 3);
            __builtin_prefetch(arow1 + kb + 256, 0, 3);
        }
        v16bf A0 = load_a_bf16(arow0, kb, lane);
        v16bf A1 = load_a_bf16(arow1, kb, lane);
#pragma unroll
        for (int j = 0; j < 4; ++j) {
            const __bf16* bp =
                Wt + (size_t)(nBase + j * 16 + (lane & 15)) * DM + kb + khSel;
            v16bf Bm = load_b_bf16c(bp);
            acc[0][j] = WMMA_BF16(A0, Bm, acc[0][j]);
            acc[1][j] = WMMA_BF16(A1, Bm, acc[1][j]);
        }
    }

    const int colLane = lane & 15;
#pragma unroll
    for (int mi = 0; mi < 2; ++mi) {
        const int rowHalf = rowBase + mi * 16 + ((lane & 16) ? 8 : 0);
#pragma unroll
        for (int j = 0; j < 4; ++j) {
            const int n = nBase + j * 16 + colLane;
            const int h = n >> 6, d = n & 63;
#pragma unroll
            for (int r = 0; r < 8; ++r) {
                const int m = rowHalf + r;
                const int b = m >> 11, s = m & 2047;
                const float v = acc[mi][j][r];
                if (which == 0)
                    qs[(((size_t)b * H_ + h) * S_ + s) * DK + d] = (__bf16)(v * 0.125f);
                else if (which == 1)
                    kk[(((size_t)b * H_ + h) * S_ + s) * DK + d] = (__bf16)v;
                else
                    vt[(((size_t)b * H_ + h) * DK + d) * S_ + s] = (__bf16)v;
            }
        }
    }
}

// ---------------------------------------------------------------------------
// Kernel 2: relative-position bucket LUT.  lut[h][rel+2047] = rel_bias[bucket,h]
// Bidirectional T5 bucketing: nb=16 (halved), max_exact=8, max_dist=128.
// ---------------------------------------------------------------------------
__global__ void bias_lut_kernel(const float* __restrict__ rel_bias,
                                float* __restrict__ lut) {
    const int idx = blockIdx.x * blockDim.x + threadIdx.x;
    if (idx >= 4095 * H_) return;
    const int rel = idx >> 4;          // 0..4094
    const int h   = idx & 15;
    const int relpos = rel - 2047;     // k - q
    const int rb = (relpos > 0) ? 16 : 0;
    const int rp = (relpos < 0) ? -relpos : relpos;
    int bucket;
    if (rp < 8) {
        bucket = rb + rp;
    } else {
        // 8 + log(rp/8)/log(16)*8, truncated, capped at 15
        float lg = __logf((float)rp * 0.125f) * (8.0f / 2.772588722f);
        int large = 8 + (int)lg;
        if (large > 15) large = 15;
        bucket = rb + large;
    }
    lut[h * LUTW + rel] = rel_bias[bucket * H_ + h];
}

// ---------------------------------------------------------------------------
// Kernel 3: position_bias output writer [B,H,S,S] = lut[h][k-q+2047] + mask[b][k]
// Pure bandwidth (1.07 GB); float4 stores.
// ---------------------------------------------------------------------------
__global__ __launch_bounds__(512)
void bias_out_kernel(const float* __restrict__ lut,
                     const float* __restrict__ mask,
                     float* __restrict__ outb) {
    const int q = blockIdx.x, h = blockIdx.y, b = blockIdx.z;
    const int k0 = threadIdx.x * 4;                       // 512 thr * 4 = 2048
    const float* lrow = lut + h * LUTW + (k0 - q + 2047);
    const float* mrow = mask + b * S_ + k0;
    float4 v;
    v.x = lrow[0] + mrow[0];
    v.y = lrow[1] + mrow[1];
    v.z = lrow[2] + mrow[2];
    v.w = lrow[3] + mrow[3];
    const size_t off = ((((size_t)b * H_ + h) * S_ + q) * S_ + k0);
    *(float4*)(outb + off) = v;
}

// ---------------------------------------------------------------------------
// Kernel 4: flash attention per (b, h, 64-row q-tile); 4 waves, 16 q-rows each.
// scores = (Q/8)·K^T + bias ; online softmax ; O += P·V ; bf16 WMMA throughout.
// ---------------------------------------------------------------------------
__global__ __launch_bounds__(128)
void attn_kernel(const __bf16* __restrict__ qs,
                 const __bf16* __restrict__ kk,
                 const __bf16* __restrict__ vt,
                 const float*  __restrict__ lut,
                 const float*  __restrict__ mask,
                 __bf16* __restrict__ attn_ws) {
    __shared__ float s_lut[LUTW];
    __shared__ __align__(16) __bf16 s_p[4][16 * 64];

    const int lane = threadIdx.x & 31;
    const int wave = threadIdx.x >> 5;
    const int qb   = blockIdx.x * 64;
    const int h    = blockIdx.y;
    const int b    = blockIdx.z;

    for (int i = threadIdx.x; i < 4095; i += 128) s_lut[i] = lut[h * LUTW + i];
    __syncthreads();

    const __bf16* qbase = qs + (((size_t)b * H_ + h) * S_) * DK;
    const __bf16* kbase = kk + (((size_t)b * H_ + h) * S_) * DK;
    const __bf16* vbase = vt + (((size_t)b * H_ + h) * DK) * S_;
    const float*  mrow  = mask + b * S_;

    const int colLane = lane & 15;
    const int rHalf   = (lane & 16) ? 8 : 0;
    const int qRowC   = qb + wave * 16 + rHalf;   // C-layout row base

    // Pre-load Q A-fragments (contraction D=64 -> two 32-wide K steps)
    const __bf16* qrow = qbase + (size_t)(qb + wave * 16 + colLane) * DK;
    v16bf Aq0 = load_a_bf16(qrow, 0, lane);
    v16bf Aq1 = load_a_bf16(qrow, 32, lane);

    float mrun[8], lrun[8];
    v8f o[4] = {};
#pragma unroll
    for (int r = 0; r < 8; ++r) { mrun[r] = -1e30f; lrun[r] = 0.0f; }

    for (int kv = 0; kv < S_; kv += 64) {
        if (kv + 64 < S_) {
            __builtin_prefetch(kbase + (size_t)(kv + 64 + colLane) * DK, 0, 2);
            __builtin_prefetch(vbase + (size_t)colLane * S_ + kv + 64, 0, 2);
        }
        // ---- scores = Q·K^T (16x64 per wave) ----
        v8f sc[4] = {};
#pragma unroll
        for (int j = 0; j < 4; ++j) {
            const int n = kv + j * 16 + colLane;         // kv position (B col)
            const __bf16* kr = kbase + (size_t)n * DK;
            const int khA = (lane & 16) ? 16 : 0;
            v16bf B0 = load_b_bf16c(kr + khA);           // D = 0..31 half
            sc[j] = WMMA_BF16(Aq0, B0, sc[j]);
            v16bf B1 = load_b_bf16c(kr + 32 + khA);      // D = 32..63 half
            sc[j] = WMMA_BF16(Aq1, B1, sc[j]);
        }

        // ---- + bias, tile row-max ----
        float tmax[8];
#pragma unroll
        for (int r = 0; r < 8; ++r) tmax[r] = -1e30f;
#pragma unroll
        for (int j = 0; j < 4; ++j) {
            const int k = kv + j * 16 + colLane;
            const float mk = mrow[k];
#pragma unroll
            for (int r = 0; r < 8; ++r) {
                const int q = qRowC + r;
                float v = sc[j][r] + s_lut[k - q + 2047] + mk;
                sc[j][r] = v;
                tmax[r] = fmaxf(tmax[r], v);
            }
        }
        // row-max across the 16 lanes sharing each row set (xor masks < 16)
#pragma unroll
        for (int r = 0; r < 8; ++r) {
            float v = tmax[r];
            v = fmaxf(v, __shfl_xor(v, 1, 32));
            v = fmaxf(v, __shfl_xor(v, 2, 32));
            v = fmaxf(v, __shfl_xor(v, 4, 32));
            v = fmaxf(v, __shfl_xor(v, 8, 32));
            tmax[r] = v;
        }

        float fscale[8], tsum[8];
#pragma unroll
        for (int r = 0; r < 8; ++r) {
            const float mnew = fmaxf(mrun[r], tmax[r]);
            fscale[r] = __expf(mrun[r] - mnew);
            mrun[r]   = mnew;
            tsum[r]   = 0.0f;
        }
#pragma unroll
        for (int j = 0; j < 4; ++j)
#pragma unroll
            for (int r = 0; r < 8; ++r) {
                const float p = __expf(sc[j][r] - mrun[r]);
                sc[j][r] = p;
                tsum[r] += p;
            }
#pragma unroll
        for (int r = 0; r < 8; ++r) {
            float v = tsum[r];
            v += __shfl_xor(v, 1, 32);
            v += __shfl_xor(v, 2, 32);
            v += __shfl_xor(v, 4, 32);
            v += __shfl_xor(v, 8, 32);
            lrun[r] = lrun[r] * fscale[r] + v;
        }
#pragma unroll
        for (int j = 0; j < 4; ++j)
#pragma unroll
            for (int r = 0; r < 8; ++r) o[j][r] *= fscale[r];

        // ---- stage P (C layout) -> LDS -> reload as A fragments ----
        __syncthreads();
#pragma unroll
        for (int j = 0; j < 4; ++j)
#pragma unroll
            for (int r = 0; r < 8; ++r)
                s_p[wave][(rHalf + r) * 64 + j * 16 + colLane] = (__bf16)sc[j][r];
        __syncthreads();

        // ---- O += P·V (contraction over kv tile of 64 -> two 32-K steps) ----
        const __bf16* prow = &s_p[wave][colLane * 64];
#pragma unroll
        for (int ks = 0; ks < 2; ++ks) {
            v16bf Ap = load_a_bf16(prow, ks * 32, lane);
            const int kh = kv + ks * 32 + ((lane & 16) ? 16 : 0);
#pragma unroll
            for (int j = 0; j < 4; ++j) {
                const int d = j * 16 + colLane;
                v16bf Bv = load_b_bf16c(vbase + (size_t)d * S_ + kh);
                o[j] = WMMA_BF16(Ap, Bv, o[j]);
            }
        }
    }

    // ---- normalize, store O as bf16 in [B, S, H*DK] for the output GEMM ----
    float rinv[8];
#pragma unroll
    for (int r = 0; r < 8; ++r) rinv[r] = 1.0f / lrun[r];
#pragma unroll
    for (int j = 0; j < 4; ++j) {
        const int d = j * 16 + colLane;
#pragma unroll
        for (int r = 0; r < 8; ++r) {
            const int q = qRowC + r;
            attn_ws[((size_t)b * S_ + q) * DM + h * DK + d] = (__bf16)(o[j][r] * rinv[r]);
        }
    }
}

// ---------------------------------------------------------------------------
// Kernel 5: output projection.  out = attn(bf16)[8192,1024] @ Wot(bf16,[N][K])
// Each wave computes 32x64 like qkv_proj; f32 output.
// ---------------------------------------------------------------------------
__global__ __launch_bounds__(128)
void out_proj_kernel(const __bf16* __restrict__ Aattn,
                     const __bf16* __restrict__ Wot,
                     float* __restrict__ out) {
    const int lane = threadIdx.x & 31;
    const int wave = threadIdx.x >> 5;
    const int rowBase = blockIdx.x * 128 + wave * 32;
    const int nBase   = blockIdx.y * 64;
    const int khSel   = (lane & 16) ? 16 : 0;

    v8f acc[2][4] = {};
    const __bf16* arow0 = Aattn + (size_t)(rowBase + (lane & 15)) * DM;
    const __bf16* arow1 = arow0 + (size_t)16 * DM;

    for (int kb = 0; kb < DM; kb += 32) {
        if (kb + 256 < DM) {
            __builtin_prefetch(arow0 + kb + 256, 0, 3);
            __builtin_prefetch(arow1 + kb + 256, 0, 3);
        }
        v16bf A0 = load_a_bf16(arow0, kb, lane);
        v16bf A1 = load_a_bf16(arow1, kb, lane);
#pragma unroll
        for (int j = 0; j < 4; ++j) {
            const __bf16* bp =
                Wot + (size_t)(nBase + j * 16 + (lane & 15)) * DM + kb + khSel;
            v16bf Bm = load_b_bf16c(bp);
            acc[0][j] = WMMA_BF16(A0, Bm, acc[0][j]);
            acc[1][j] = WMMA_BF16(A1, Bm, acc[1][j]);
        }
    }

    const int colLane = lane & 15;
#pragma unroll
    for (int mi = 0; mi < 2; ++mi) {
        const int rowHalf = rowBase + mi * 16 + ((lane & 16) ? 8 : 0);
#pragma unroll
        for (int j = 0; j < 4; ++j) {
            const int n = nBase + j * 16 + colLane;
#pragma unroll
            for (int r = 0; r < 8; ++r)
                out[(size_t)(rowHalf + r) * DM + n] = acc[mi][j][r];
        }
    }
}

// ---------------------------------------------------------------------------
extern "C" void kernel_launch(void* const* d_in, const int* in_sizes, int n_in,
                              void* d_out, int out_size, void* d_ws, size_t ws_size,
                              hipStream_t stream) {
    const float* hs       = (const float*)d_in[0];  // [B,S,DM]
    const float* mask     = (const float*)d_in[1];  // [B,S]
    const float* wq       = (const float*)d_in[2];  // [DM, H*DK]
    const float* wk       = (const float*)d_in[3];
    const float* wv       = (const float*)d_in[4];
    const float* wo       = (const float*)d_in[5];  // [H*DK, DM]
    const float* rel_bias = (const float*)d_in[6];  // [32, H]

    char* ws = (char*)d_ws;
    // xb is dead after qkv_proj; attn output aliases it.
    __bf16* xb   = (__bf16*)(ws);                        // 16 MB [B*S, DM]
    __bf16* attn = (__bf16*)(ws);                        //        (alias)
    __bf16* qs   = (__bf16*)(ws + (size_t)(16 << 20));   // 16 MB [B,H,S,DK]
    __bf16* kk   = (__bf16*)(ws + (size_t)(32 << 20));   // 16 MB [B,H,S,DK]
    __bf16* vt   = (__bf16*)(ws + (size_t)(48 << 20));   // 16 MB [B,H,DK,S]
    __bf16* wqt  = (__bf16*)(ws + (size_t)(64 << 20));   // 2 MB  [N][K]
    __bf16* wkt  = (__bf16*)(ws + (size_t)(66 << 20));   // 2 MB
    __bf16* wvt  = (__bf16*)(ws + (size_t)(68 << 20));   // 2 MB
    __bf16* wot  = (__bf16*)(ws + (size_t)(70 << 20));   // 2 MB
    float*  lut  = (float*) (ws + (size_t)(72 << 20));   // 256 KB [H,4096]

    float* out_attn = (float*)d_out;                     // [B,S,DM]
    float* out_bias = out_attn + (size_t)B_ * S_ * DM;   // [B,H,S,S]

    // 0) one-time bf16 conversion of activations + transposed weights
    cvt_x_kernel<<<(B_ * S_ * DM) / (256 * 8), 256, 0, stream>>>(hs, xb);
    cvt_wt_kernel<<<dim3(DM / 32, DM / 32, 4), 256, 0, stream>>>(
        wq, wk, wv, wo, wqt, wkt, wvt, wot);

    // 1) fused QKV projections (z = 0/1/2 -> Q/K/V)
    qkv_proj_kernel<<<dim3(B_ * S_ / 128, DM / 64, 3), 128, 0, stream>>>(
        xb, wqt, wkt, wvt, qs, kk, vt);

    // 2) relative-position bucket LUT
    bias_lut_kernel<<<(4095 * H_ + 255) / 256, 256, 0, stream>>>(rel_bias, lut);

    // 3) position_bias output (pure bandwidth)
    bias_out_kernel<<<dim3(S_, H_, B_), 512, 0, stream>>>(lut, mask, out_bias);

    // 4) flash attention (writes attn, which aliases the now-dead xb)
    attn_kernel<<<dim3(S_ / 64, H_, B_), 128, 0, stream>>>(
        qs, kk, vt, lut, mask, attn);

    // 5) output projection
    out_proj_kernel<<<dim3(B_ * S_ / 128, DM / 64, 1), 128, 0, stream>>>(
        attn, wot, out_attn);
}
